// TimeMixing_55216099557850
// MI455X (gfx1250) — matrix operand order repaired and checked by reading, not compiled, not run on hardware
//
#include <hip/hip_runtime.h>

// ---------------------------------------------------------------------------
// RWKV time-mixing for MI455X (gfx1250, wave32, WMMA + Tensor Data Mover).
// 4 GEMMs (550 GFLOP) via v_wmma_f32_16x16x32_bf16; A/W tiles staged into LDS
// by TDM (tensor_load_to_lds, TENSORcnt), double-buffered; sequential WKV scan.
// ---------------------------------------------------------------------------

typedef __bf16 bf16_t;
typedef __attribute__((ext_vector_type(16))) __bf16 v16bf;
typedef __attribute__((ext_vector_type(8)))  __bf16 v8bf;
typedef __attribute__((ext_vector_type(8)))  float  v8f;
typedef __attribute__((ext_vector_type(4)))  unsigned int v4u;
typedef __attribute__((ext_vector_type(8)))  int v8i;
typedef __attribute__((ext_vector_type(4)))  int v4i;

static constexpr int Bsz = 4;
static constexpr int Tsz = 4096;
static constexpr int Dsz = 2048;

#ifndef __has_builtin
#define __has_builtin(x) 0
#endif
#if __has_builtin(__builtin_amdgcn_tensor_load_to_lds) && \
    __has_builtin(__builtin_amdgcn_s_wait_tensorcnt)
#define HAVE_TDM 1
#else
#define HAVE_TDM 0
#endif

// ------------------------- fp32 -> bf16 weight convert ---------------------
__global__ __launch_bounds__(256) void cvt_bf16(const float* __restrict__ s,
                                                bf16_t* __restrict__ d) {
  size_t i = (size_t)blockIdx.x * 256 + threadIdx.x;
  d[i] = (bf16_t)s[i];
}

// --------------------- time-shift + token mixes (bf16 out) -----------------
__global__ __launch_bounds__(256) void prep_mix(const float* __restrict__ x,
                                                const float* __restrict__ tmk,
                                                const float* __restrict__ tmv,
                                                const float* __restrict__ tmr,
                                                bf16_t* __restrict__ ak,
                                                bf16_t* __restrict__ av,
                                                bf16_t* __restrict__ ar) {
  size_t i = (size_t)blockIdx.x * 256 + threadIdx.x;   // over B*T*D
  int d = (int)(i & (Dsz - 1));
  size_t bt = i >> 11;                                  // D = 2^11
  int t = (int)(bt & (Tsz - 1));                        // T = 2^12
  float xv = x[i];
  float xs = (t > 0) ? x[i - Dsz] : 0.0f;
  float mk = tmk[d], mv = tmv[d], mr = tmr[d];
  ak[i] = (bf16_t)(xv * mk + xs * (1.0f - mk));
  av[i] = (bf16_t)(xv * mv + xs * (1.0f - mv));
  float rr = xv * mr + xs * (1.0f - mr);
  ar[i] = (bf16_t)(rr / (1.0f + __expf(-rr)));          // SiLU
}

// ------------------- TDM: DMA one 128 x 64-bf16 tile into LDS --------------
// Tile rows are 128 B (32 dwords); D# pad adds 16 B (4 dwords) after each row
// => LDS row stride 144 B = 72 bf16 (matches the fragment layout below).
#if HAVE_TDM
__device__ __forceinline__ void tdm_tile_load(unsigned lds_addr,
                                              const bf16_t* gptr, int K) {
  unsigned long long ga = (unsigned long long)(uintptr_t)gptr;
  const unsigned rowUnits = 32;                  // tile_dim0: 32 x 4B = 64 bf16
  const unsigned rows = 128;                     // tile_dim1
  unsigned long long strideU = (unsigned long long)(K / 2);  // row stride, 4B units

  v4u g0;
  g0.x = 1u;                                            // count=1 (user D#)
  g0.y = lds_addr;                                      // LDS byte address
  g0.z = (unsigned)ga;                                  // global_addr[31:0]
  g0.w = (unsigned)((ga >> 32) & 0x01FFFFFFu) | 0x80000000u;  // [56:32] | type=2

  v8i g1;
  g1[0] = (int)((2u << 16)      // data_size = 4B
              | (1u << 20)      // pad_enable
              | (4u << 22)      // pad_interval code 4 => every 32 dwords (128B)
              | (3u << 25));    // pad_amount   code 3 => 4 dwords (16B)
  g1[1] = (int)((rowUnits & 0xFFFFu) << 16);            // tensor_dim0 lo16 (=32)
  g1[2] = (int)((rows & 0xFFFFu) << 16);                // dim0 hi=0 | tensor_dim1 lo (=128)
  g1[3] = (int)((rowUnits & 0xFFFFu) << 16);            // dim1 hi=0 | tile_dim0 (=32)
  g1[4] = (int)(rows & 0xFFFFu);                        // tile_dim1=128, tile_dim2=0
  g1[5] = (int)(unsigned)(strideU & 0xFFFFFFFFu);       // tensor_dim0_stride[31:0]
  g1[6] = (int)(unsigned)((strideU >> 32) & 0xFFFFu);   // stride[47:32] | dim1_stride lo=0
  g1[7] = 0;

  v4i z = {0, 0, 0, 0};
#if __clang_major__ >= 23
  v8i z8 = {0, 0, 0, 0, 0, 0, 0, 0};
  __builtin_amdgcn_tensor_load_to_lds(g0, g1, z, z, z8, 0);
#else
  __builtin_amdgcn_tensor_load_to_lds(g0, g1, z, z, 0);
#endif
}
#endif

// --------------------------- WMMA bf16 GEMM --------------------------------
// C[M,N] = A[M,K] @ W[N,K]^T + bias  (nn.Linear semantics: both K-contiguous)
// Block = 256 thr = 8 waves (4 x 2). Wave tile 32x64 => 2x4 16x16 accums.
// TDM double-buffers 128x64 A/W slabs in LDS while waves run the matrix pipes.
template <bool OUT_BF16>
__global__ __launch_bounds__(256) void gemm_bf16(const bf16_t* __restrict__ A,
                                                 const bf16_t* __restrict__ W,
                                                 const float* __restrict__ bias,
                                                 void* __restrict__ outv,
                                                 int M, int N, int K) {
  constexpr int TM = 128, TN = 128, TK = 64, LS = 72;  // LS: padded LDS row stride
  __shared__ bf16_t Als[2 * TM * LS];
  __shared__ bf16_t Wls[2 * TN * LS];

  const int tid  = threadIdx.x;
  const int lane = tid & 31;
  const int wave = tid >> 5;
  const int wm   = wave >> 1;        // 0..3  (M direction)
  const int wn   = wave & 1;         // 0..1  (N direction)
  const int l15  = lane & 15;
  const int lh   = lane >> 4;        // lane half (ISA bf16 fragment layout)
  const int tile_n = blockIdx.x * TN;
  const int tile_m = blockIdx.y * TM;
  const int nstages = K / TK;

  v8f acc[2][4] = {};

#if HAVE_TDM
  if (wave == 0) {                   // one wave drives the Tensor Data Mover
    tdm_tile_load((unsigned)(uintptr_t)&Als[0], A + (size_t)tile_m * K, K);
    tdm_tile_load((unsigned)(uintptr_t)&Wls[0], W + (size_t)tile_n * K, K);
  }
#endif

  for (int s = 0; s < nstages; ++s) {
    const int kt   = s * TK;
    const int bufo = (s & 1) * TM * LS;

#if HAVE_TDM
    if (wave == 0) {
      if (s + 1 < nstages) {         // kick next slab, then wait for current pair
        const int nb = ((s + 1) & 1) * TM * LS;
        tdm_tile_load((unsigned)(uintptr_t)&Als[nb],
                      A + (size_t)tile_m * K + kt + TK, K);
        tdm_tile_load((unsigned)(uintptr_t)&Wls[nb],
                      W + (size_t)tile_n * K + kt + TK, K);
        __builtin_amdgcn_s_wait_tensorcnt((short)2);
      } else {
        __builtin_amdgcn_s_wait_tensorcnt((short)0);
      }
    }
    __syncthreads();                 // current buffer visible to all waves
#else
    for (int c = tid; c < TM * TK / 8; c += 256) {
      int r  = c >> 3;
      int k8 = (c & 7) << 3;
      *(uint4*)&Als[bufo + r * LS + k8] =
          *(const uint4*)&A[(size_t)(tile_m + r) * K + kt + k8];
      *(uint4*)&Wls[bufo + r * LS + k8] =
          *(const uint4*)&W[(size_t)(tile_n + r) * K + kt + k8];
    }
    __syncthreads();
#endif

    #pragma unroll
    for (int ks = 0; ks < TK; ks += 32) {
      union FR { v16bf v; v8bf h[2]; };
      FR af[2], bfr[4];
      // A fragment (16x32 bf16): lane half selects K {0..7,16..23} vs {8..15,24..31}
      #pragma unroll
      for (int i = 0; i < 2; ++i) {
        int row = wm * 32 + i * 16 + l15;
        int kA  = ks + lh * 8;
        af[i].h[0] = *(const v8bf*)&Als[bufo + row * LS + kA];
        af[i].h[1] = *(const v8bf*)&Als[bufo + row * LS + kA + 16];
      }
      // B fragment (32x16 bf16): lane half selects K {0..7,16..23} vs {8..15,24..31}
      #pragma unroll
      for (int j = 0; j < 4; ++j) {
        int col = wn * 64 + j * 16 + l15;
        int kB  = ks + lh * 8;
        bfr[j].h[0] = *(const v8bf*)&Wls[bufo + col * LS + kB];
        bfr[j].h[1] = *(const v8bf*)&Wls[bufo + col * LS + kB + 16];
      }
      #pragma unroll
      for (int i = 0; i < 2; ++i)
        #pragma unroll
        for (int j = 0; j < 4; ++j)
          acc[i][j] = __builtin_amdgcn_wmma_f32_16x16x32_bf16(
              false, af[i].v, false, bfr[j].v, (short)0, acc[i][j], false, false);
    }
    __syncthreads();                 // done reading: buffer may be re-filled
  }

  // ---- epilogue: C layout = VGPR vg -> M = vg + 8*laneHalf, N = lane&15 ----
  #pragma unroll
  for (int i = 0; i < 2; ++i) {
    #pragma unroll
    for (int j = 0; j < 4; ++j) {
      int col  = tile_n + wn * 64 + j * 16 + l15;
      float bv = bias[col];
      #pragma unroll
      for (int vg = 0; vg < 8; ++vg) {
        int row = tile_m + wm * 32 + i * 16 + lh * 8 + vg;
        float val = acc[i][j][vg] + bv;
        if constexpr (OUT_BF16)
          ((bf16_t*)outv)[(size_t)row * N + col] = (bf16_t)val;
        else
          ((float*)outv)[(size_t)row * N + col] = val;
      }
    }
  }
}

// ------------------------------- WKV scan ----------------------------------
// One thread per (b,d) channel, serial over T. Writes wkv*r in place over k.
__global__ __launch_bounds__(256) void wkv_scan(bf16_t* __restrict__ kb,
                                                const bf16_t* __restrict__ vb,
                                                const bf16_t* __restrict__ rb,
                                                const float* __restrict__ u,
                                                const float* __restrict__ w) {
  int idx = blockIdx.x * 256 + threadIdx.x;   // 0 .. B*D-1
  int b = idx >> 11;
  int d = idx & (Dsz - 1);
  float uu = u[d], ww = w[d];
  float aa = 0.0f, bq = 0.0f;
  size_t base = (size_t)b * Tsz * Dsz + d;
  for (int t = 0; t < Tsz; ++t) {
    size_t p = base + (size_t)t * Dsz;
    float kt = (float)kb[p];
    float vt = (float)vb[p];
    float rt = (float)rb[p];
    float tm = fmaxf(uu + kt, ww);
    float e1 = __expf(-ww - tm);
    float e2 = __expf(uu + kt - tm);
    aa = e1 * aa + e2 * vt;
    bq = e1 * bq + e2;
    kb[p] = (bf16_t)((aa / bq) * rt);         // wkv * r, in place (read-then-write)
  }
}

// ------------------------------- launch ------------------------------------
extern "C" void kernel_launch(void* const* d_in, const int* in_sizes, int n_in,
                              void* d_out, int out_size, void* d_ws, size_t ws_size,
                              hipStream_t stream) {
  (void)in_sizes; (void)n_in; (void)out_size; (void)ws_size;
  const float* x   = (const float*)d_in[0];
  const float* Wk  = (const float*)d_in[1];
  const float* bk  = (const float*)d_in[2];
  const float* Wv  = (const float*)d_in[3];
  const float* bv  = (const float*)d_in[4];
  const float* Wr  = (const float*)d_in[5];
  const float* br  = (const float*)d_in[6];
  const float* Wo  = (const float*)d_in[7];
  const float* bo  = (const float*)d_in[8];
  const float* tmk = (const float*)d_in[9];
  const float* tmv = (const float*)d_in[10];
  const float* tmr = (const float*)d_in[11];
  const float* u   = (const float*)d_in[12];
  const float* w   = (const float*)d_in[13];

  const size_t nBTD = (size_t)Bsz * Tsz * Dsz;   // 33.5M
  const size_t nDD  = (size_t)Dsz * Dsz;         // 4.2M

  bf16_t* p   = (bf16_t*)d_ws;
  bf16_t* ak  = p; p += nBTD;
  bf16_t* av  = p; p += nBTD;
  bf16_t* ar  = p; p += nBTD;
  bf16_t* wkb = p; p += nDD;
  bf16_t* wvb = p; p += nDD;
  bf16_t* wrb = p; p += nDD;
  bf16_t* wob = p; p += nDD;
  bf16_t* kb  = p; p += nBTD;   // k; later overwritten with wkv*r
  bf16_t* vbp = p; p += nBTD;
  bf16_t* rbp = p; p += nBTD;

  cvt_bf16<<<(unsigned)(nDD / 256), 256, 0, stream>>>(Wk, wkb);
  cvt_bf16<<<(unsigned)(nDD / 256), 256, 0, stream>>>(Wv, wvb);
  cvt_bf16<<<(unsigned)(nDD / 256), 256, 0, stream>>>(Wr, wrb);
  cvt_bf16<<<(unsigned)(nDD / 256), 256, 0, stream>>>(Wo, wob);
  prep_mix<<<(unsigned)(nBTD / 256), 256, 0, stream>>>(x, tmk, tmv, tmr, ak, av, ar);

  dim3 gg(Dsz / 128, (Bsz * Tsz) / 128);   // (16, 128) tiles
  gemm_bf16<true ><<<gg, 256, 0, stream>>>(ak, wkb, bk, kb,  Bsz * Tsz, Dsz, Dsz);
  gemm_bf16<true ><<<gg, 256, 0, stream>>>(av, wvb, bv, vbp, Bsz * Tsz, Dsz, Dsz);
  gemm_bf16<true ><<<gg, 256, 0, stream>>>(ar, wrb, br, rbp, Bsz * Tsz, Dsz, Dsz);

  wkv_scan<<<(Bsz * Dsz) / 256, 256, 0, stream>>>(kb, vbp, rbp, u, w);

  gemm_bf16<false><<<gg, 256, 0, stream>>>(kb, wob, bo, d_out, Bsz * Tsz, Dsz, Dsz);
}
